// Attention_84129819394524
// MI455X (gfx1250) — matrix operand order repaired and checked by reading, not compiled, not run on hardware
//
#include <hip/hip_runtime.h>

#define NHEAD 12
#define DH    64
#define CDIM  768
#define NB    4
#define NSEQ  1024
#define C3    (3 * CDIM)
#define QK_SCALE 0.125f   // 64^-0.5

typedef __attribute__((ext_vector_type(16))) _Float16 v16h;
typedef __attribute__((ext_vector_type(8)))  _Float16 v8h;
typedef __attribute__((ext_vector_type(8)))  float    v8f;
typedef __attribute__((ext_vector_type(4)))  float    v4f;

static __device__ __forceinline__ v8f wmma16x16x32(v16h a, v16h b, v8f c) {
    // (neg_a, A, neg_b, B, c_mod, C, reuse_a, reuse_b)
    return __builtin_amdgcn_wmma_f32_16x16x32_f16(false, a, false, b, (short)0, c,
                                                  false, false);
}

// ---- fragment builders (CDNA5 wave32 WMMA layouts, CDNA5 ISA) ----
// A 16x32 f16: lane m (m<16) holds K {0..7, 16..23}; lane m+16 holds {8..15, 24..31}
// B 32x16 f16: lane n (n<16) holds K {0..7, 16..23} of col n; lane n+16 holds {8..15, 24..31}

static __device__ __forceinline__ v16h a_frag_pack(v8h lo, v8h hi) {
    v16h out;
#pragma unroll
    for (int i = 0; i < 8; ++i) { out[i] = lo[i]; out[i + 8] = hi[i]; }
    return out;
}

static __device__ __forceinline__ v16h a_frag_h16(const _Float16* base, int ld,
                                                  int row_base, int kbase,
                                                  int r16, int seg) {
    const _Float16* p = base + (long)(row_base + r16) * ld + kbase + seg * 8;
    return a_frag_pack(*(const v8h*)p, *(const v8h*)(p + 16));
}

static __device__ __forceinline__ v16h b_frag_h16(const _Float16* base, int ld,
                                                  int col_base, int kbase,
                                                  int r16, int seg) {
    const _Float16* p = base + (long)(col_base + r16) * ld + kbase + seg * 8;
    return a_frag_pack(*(const v8h*)p, *(const v8h*)(p + 16));
}

// ============================================================================
// Kernel 0: f32 -> f16 convert prepass (x, w_qkv, w_proj). Memory-bound,
// ~33 MB total traffic; makes every GEMM inner loop pure-f16.
// ============================================================================
__global__ __launch_bounds__(256) void cvt_f16_kernel(
        const float* __restrict__ src, _Float16* __restrict__ dst, long n) {
    const long i = ((long)blockIdx.x * blockDim.x + threadIdx.x) * 8;
    if (i + 8 <= n) {
        v4f a = *(const v4f*)(src + i);
        v4f b = *(const v4f*)(src + i + 4);
        v8h o;
#pragma unroll
        for (int j = 0; j < 4; ++j) {
            o[j]     = (_Float16)a[j];
            o[j + 4] = (_Float16)b[j];
        }
        *(v8h*)(dst + i) = o;
    }
}

// ============================================================================
// GEMM core: one wave computes a 32x64 tile of C = A @ W^T (both f16 [*,768]
// row-major). K fully unrolled (compile-time 768) -> k offsets become load
// immediates; manual 2-deep ping-pong (no loop-carried register rotation).
// ============================================================================
static __device__ __forceinline__ void gemm_tile_32x64(
        const _Float16* __restrict__ A, const _Float16* __restrict__ W,
        int row0, int col0, int r16, int seg, v8f acc[2][4]) {
    const _Float16* ar0 = A + (long)(row0 + r16) * CDIM + seg * 8;
    const _Float16* ar1 = ar0 + 16 * CDIM;
    const _Float16* wr0 = W + (long)(col0 + r16) * CDIM + seg * 8;

    v16h a0[2], a1[2], b0[4], b1[4];
    a0[0] = a_frag_pack(*(const v8h*)ar0, *(const v8h*)(ar0 + 16));
    a0[1] = a_frag_pack(*(const v8h*)ar1, *(const v8h*)(ar1 + 16));
#pragma unroll
    for (int j = 0; j < 4; ++j) b0[j] = a_frag_pack(*(const v8h*)(wr0 + (long)j * 16 * CDIM), *(const v8h*)(wr0 + (long)j * 16 * CDIM + 16));

#pragma unroll
    for (int k = 0; k < CDIM; k += 64) {
        // stage k+32 while WMMAs for k issue
        a1[0] = a_frag_pack(*(const v8h*)(ar0 + k + 32), *(const v8h*)(ar0 + k + 48));
        a1[1] = a_frag_pack(*(const v8h*)(ar1 + k + 32), *(const v8h*)(ar1 + k + 48));
#pragma unroll
        for (int j = 0; j < 4; ++j)
            b1[j] = a_frag_pack(*(const v8h*)(wr0 + (long)j * 16 * CDIM + k + 32), *(const v8h*)(wr0 + (long)j * 16 * CDIM + k + 48));
#pragma unroll
        for (int j = 0; j < 4; ++j) {
            acc[0][j] = wmma16x16x32(a0[0], b0[j], acc[0][j]);
            acc[1][j] = wmma16x16x32(a0[1], b0[j], acc[1][j]);
        }
        // stage k+64 while WMMAs for k+32 issue
        if (k + 64 < CDIM) {
            a0[0] = a_frag_pack(*(const v8h*)(ar0 + k + 64), *(const v8h*)(ar0 + k + 80));
            a0[1] = a_frag_pack(*(const v8h*)(ar1 + k + 64), *(const v8h*)(ar1 + k + 80));
#pragma unroll
            for (int j = 0; j < 4; ++j)
                b0[j] = a_frag_pack(*(const v8h*)(wr0 + (long)j * 16 * CDIM + k + 64), *(const v8h*)(wr0 + (long)j * 16 * CDIM + k + 80));
        }
#pragma unroll
        for (int j = 0; j < 4; ++j) {
            acc[0][j] = wmma16x16x32(a1[0], b1[j], acc[0][j]);
            acc[1][j] = wmma16x16x32(a1[1], b1[j], acc[1][j]);
        }
    }
}

// ============================================================================
// Kernel 1: qkv = x @ w_qkv^T ; route to q (scaled, [BH,N,Dh]), k ([BH,N,Dh]),
//           v^T ([BH,Dh,N]) as f16. grid = (4096/32, 2304/64), block = 32.
// ============================================================================
__global__ __launch_bounds__(32) void qkv_gemm_kernel(
        const _Float16* __restrict__ xh, const _Float16* __restrict__ wh,
        _Float16* __restrict__ qb, _Float16* __restrict__ kb,
        _Float16* __restrict__ vt) {
    const int lane = threadIdx.x & 31;
    const int r16 = lane & 15, seg = lane >> 4;
    const int row0 = blockIdx.x * 32;       // over B*N = 4096
    const int col0 = blockIdx.y * 64;       // over 3C = 2304

    v8f acc[2][4] = {};
    gemm_tile_32x64(xh, wh, row0, col0, r16, seg, acc);

#pragma unroll
    for (int j = 0; j < 4; ++j) {
        const int col = col0 + j * 16 + r16;  // tile never straddles q/k/v or heads
        const int sel = col / CDIM;           // 0=q 1=k 2=v
        const int cc  = col % CDIM;
        const int h   = cc / DH, d = cc % DH;
#pragma unroll
        for (int mi = 0; mi < 2; ++mi) {
#pragma unroll
            for (int r = 0; r < 8; ++r) {
                const int row = row0 + mi * 16 + r + 8 * seg;
                const int b = row >> 10, nn = row & (NSEQ - 1);
                const long bh = (long)(b * NHEAD + h);
                const float v = acc[mi][j][r];
                if (sel == 0)
                    qb[(bh * NSEQ + nn) * DH + d] = (_Float16)(v * QK_SCALE);
                else if (sel == 1)
                    kb[(bh * NSEQ + nn) * DH + d] = (_Float16)v;
                else
                    vt[(bh * DH + d) * NSEQ + nn] = (_Float16)v;
            }
        }
    }
}

// ============================================================================
// Kernel 2: per (b,h,16-row query tile): S = q k^T -> softmax -> attn out +
//           P@V. LDS: S f32 [16][1024] + P f16 [16][1024] = 96 KB dynamic.
// grid = 48*64 blocks, block = 32.
// ============================================================================
__global__ __launch_bounds__(32) void attn_kernel(
        const _Float16* __restrict__ qb, const _Float16* __restrict__ kb,
        const _Float16* __restrict__ vt, float* __restrict__ attn_out,
        _Float16* __restrict__ ob) {
    extern __shared__ char smem[];
    float*    S = (float*)smem;                              // 16*1024*4
    _Float16* P = (_Float16*)(smem + 16 * NSEQ * 4);         // 16*1024*2

    const int lane = threadIdx.x & 31;
    const int r16 = lane & 15, seg = lane >> 4;
    const int mt = blockIdx.x & 63;
    const int bh = blockIdx.x >> 6;            // 0..47
    const int b = bh / NHEAD, h = bh % NHEAD;
    const long hoff = (long)bh * NSEQ * DH;
    const int row0 = mt * 16;

    // q A-fragments (Dh = 64 => two K=32 fragments), held in registers
    const v16h aq0 = a_frag_h16(qb + hoff, DH, row0, 0,  r16, seg);
    const v16h aq1 = a_frag_h16(qb + hoff, DH, row0, 32, r16, seg);

    // ---- scores: 64 key tiles, 2 WMMAs each; 2-deep ping-pong, no rotation
    {
        v16h k0a = b_frag_h16(kb + hoff, DH, 0, 0,  r16, seg);
        v16h k0b = b_frag_h16(kb + hoff, DH, 0, 32, r16, seg);
        for (int nt = 0; nt < 64; nt += 2) {
            v16h k1a = b_frag_h16(kb + hoff, DH, (nt + 1) * 16, 0,  r16, seg);
            v16h k1b = b_frag_h16(kb + hoff, DH, (nt + 1) * 16, 32, r16, seg);
            v8f s = {};
            s = wmma16x16x32(aq0, k0a, s);
            s = wmma16x16x32(aq1, k0b, s);
#pragma unroll
            for (int r = 0; r < 8; ++r)
                S[(r + 8 * seg) * NSEQ + nt * 16 + r16] = s[r];
            if (nt + 2 < 64) {
                k0a = b_frag_h16(kb + hoff, DH, (nt + 2) * 16, 0,  r16, seg);
                k0b = b_frag_h16(kb + hoff, DH, (nt + 2) * 16, 32, r16, seg);
            }
            v8f s2 = {};
            s2 = wmma16x16x32(aq0, k1a, s2);
            s2 = wmma16x16x32(aq1, k1b, s2);
#pragma unroll
            for (int r = 0; r < 8; ++r)
                S[(r + 8 * seg) * NSEQ + (nt + 1) * 16 + r16] = s2[r];
        }
    }
    __syncthreads();

    // ---- softmax: 2 lanes per row (halves joined via shfl_xor 16) ----
    {
        const int row = r16;
        float* sr = S + row * NSEQ + seg * 512;
        float m = -3.4e38f;
        for (int i = 0; i < 512; ++i) m = fmaxf(m, sr[i]);
        m = fmaxf(m, __shfl_xor(m, 16, 32));
        float sum = 0.0f;
        for (int i = 0; i < 512; ++i) {
            float e = __expf(sr[i] - m);
            sr[i] = e;
            sum += e;
        }
        sum += __shfl_xor(sum, 16, 32);
        const float inv = 1.0f / sum;
        float* ao = attn_out + ((long)bh * NSEQ + row0 + row) * NSEQ + seg * 512;
        _Float16* pr = P + row * NSEQ + seg * 512;
        for (int i = 0; i < 512; ++i) {
            float p = sr[i] * inv;
            ao[i] = p;                    // attn output (f32)
            pr[i] = (_Float16)p;          // P for the PV matmul
        }
    }
    __syncthreads();

    // ---- out = P @ V : 4 Dh-col tiles x 32 K-steps; 2-deep ping-pong on V^T
    for (int ct = 0; ct < 4; ++ct) {
        v8f o = {};
        v16h v0 = b_frag_h16(vt + hoff, NSEQ, ct * 16, 0, r16, seg);
        for (int kt = 0; kt < 32; kt += 2) {
            v16h v1 = b_frag_h16(vt + hoff, NSEQ, ct * 16, (kt + 1) * 32, r16, seg);
            v16h ap = a_frag_h16(P, NSEQ, 0, kt * 32, r16, seg);     // LDS
            o = wmma16x16x32(ap, v0, o);
            if (kt + 2 < 32)
                v0 = b_frag_h16(vt + hoff, NSEQ, ct * 16, (kt + 2) * 32, r16, seg);
            v16h ap2 = a_frag_h16(P, NSEQ, 0, (kt + 1) * 32, r16, seg); // LDS
            o = wmma16x16x32(ap2, v1, o);
        }
#pragma unroll
        for (int r = 0; r < 8; ++r) {
            const int row = row0 + r + 8 * seg;
            ob[((long)b * NSEQ + row) * CDIM + h * DH + ct * 16 + r16] =
                (_Float16)o[r];
        }
    }
}

// ============================================================================
// Kernel 3: out = ob @ w_proj^T + b_proj  -> f32 d_out
// grid = (4096/32, 768/64), block = 32.
// ============================================================================
__global__ __launch_bounds__(32) void proj_gemm_kernel(
        const _Float16* __restrict__ ob, const _Float16* __restrict__ wh,
        const float* __restrict__ bias, float* __restrict__ out) {
    const int lane = threadIdx.x & 31;
    const int r16 = lane & 15, seg = lane >> 4;
    const int row0 = blockIdx.x * 32;
    const int col0 = blockIdx.y * 64;

    v8f acc[2][4] = {};
    gemm_tile_32x64(ob, wh, row0, col0, r16, seg, acc);

#pragma unroll
    for (int j = 0; j < 4; ++j) {
        const int col = col0 + j * 16 + r16;
        const float bv = bias[col];
#pragma unroll
        for (int mi = 0; mi < 2; ++mi) {
#pragma unroll
            for (int r = 0; r < 8; ++r) {
                const int row = row0 + mi * 16 + r + 8 * seg;
                out[(long)row * CDIM + col] = acc[mi][j][r] + bv;
            }
        }
    }
}

extern "C" void kernel_launch(void* const* d_in, const int* in_sizes, int n_in,
                              void* d_out, int out_size, void* d_ws, size_t ws_size,
                              hipStream_t stream) {
    const float* x      = (const float*)d_in[0];
    const float* w_qkv  = (const float*)d_in[1];
    const float* w_proj = (const float*)d_in[2];
    const float* b_proj = (const float*)d_in[3];

    float* out  = (float*)d_out;                              // [B,N,C]
    float* attn = out + (long)NB * NSEQ * CDIM;               // [B,H,N,N]

    // workspace layout (f16 halves)
    const long n_x   = (long)NB * NSEQ * CDIM;     // 3,145,728
    const long n_wq  = (long)C3 * CDIM;            // 1,769,472
    const long n_wp  = (long)CDIM * CDIM;          //   589,824
    const long per   = (long)NB * NHEAD * NSEQ * DH;

    _Float16* xh  = (_Float16*)d_ws;
    _Float16* wqh = xh + n_x;
    _Float16* wph = wqh + n_wq;
    _Float16* qb  = wph + n_wp;
    _Float16* kb  = qb + per;
    _Float16* vt  = kb + per;
    _Float16* ob  = vt + per;

    // prepass: convert operands to f16 once (L2-resident afterwards)
    cvt_f16_kernel<<<(unsigned)((n_x  / 8 + 255) / 256), 256, 0, stream>>>(x, xh, n_x);
    cvt_f16_kernel<<<(unsigned)((n_wq / 8 + 255) / 256), 256, 0, stream>>>(w_qkv, wqh, n_wq);
    cvt_f16_kernel<<<(unsigned)((n_wp / 8 + 255) / 256), 256, 0, stream>>>(w_proj, wph, n_wp);

    qkv_gemm_kernel<<<dim3((NB * NSEQ) / 32, C3 / 64), 32, 0, stream>>>(
        xh, wqh, qb, kb, vt);

    const size_t shmem = 16 * NSEQ * 4 + 16 * NSEQ * 2;       // 96 KB
    attn_kernel<<<dim3(NB * NHEAD * (NSEQ / 16)), 32, shmem, stream>>>(
        qb, kb, vt, attn, ob);

    proj_gemm_kernel<<<dim3((NB * NSEQ) / 32, CDIM / 64), 32, 0, stream>>>(
        ob, wph, b_proj, out);
}
